// WindowAttention_33629593928261
// MI455X (gfx1250) — matrix operand order repaired and checked by reading, not compiled, not run on hardware
//
#include <hip/hip_runtime.h>
#include <hip/hip_bf16.h>

// ---------------------------------------------------------------------------
// Types for CDNA5 WMMA (wave32)
// ---------------------------------------------------------------------------
typedef _Float16 v16h __attribute__((ext_vector_type(16)));
typedef _Float16 v8h  __attribute__((ext_vector_type(8)));
typedef float    v8f  __attribute__((ext_vector_type(8)));

#define DIM 768
#define NHEADS 24
#define HD 32
#define NTOK 49          // tokens per window
#define NWIN 1024        // 16 * 64 windows
#define MROWS (NWIN * NTOK)   // 50176 token rows
#define QKVDIM 2304

__device__ __forceinline__ v8f wmma16(v16h a, v16h b, v8f c) {
  // D = A(16x32 f16) x B(32x16 f16) + C(16x16 f32)
  return __builtin_amdgcn_wmma_f32_16x16x32_f16(
      /*neg_a=*/false, a, /*neg_b=*/false, b,
      /*c_mod=*/(short)0, c, /*reuse_a=*/false, /*reuse_b=*/false);
}

__device__ __forceinline__ v16h frag_cat(v8h a, v8h b) {
  v16h r;
#pragma unroll
  for (int i = 0; i < 8; ++i) { r[i] = a[i]; r[i + 8] = b[i]; }
  return r;
}

// A fragment (16x32 f16): lane holds row M = lane&15.
// element e: K = 16*(e>>3) + 8*(lane>>4) + (e&7)  -> two 16B LDS loads.
__device__ __forceinline__ v16h load_fragA(const _Float16* base, int lda, int lane) {
  const _Float16* p = base + (size_t)(lane & 15) * lda + ((lane >> 4) << 3);
  v8h lo = *(const v8h*)p;
  v8h hi = *(const v8h*)(p + 16);
  return frag_cat(lo, hi);
}

// B fragment (32x16 f16): lane holds col N = lane&15.
// element e: K = 16*(e>>3) + 8*(lane>>4) + (e&7)  -> two 16B LDS loads.
__device__ __forceinline__ v16h load_fragB(const _Float16* base, int lda, int lane) {
  const _Float16* p = base + (size_t)(lane & 15) * lda + ((lane >> 4) << 3);
  v8h lo = *(const v8h*)p;
  v8h hi = *(const v8h*)(p + 16);
  return frag_cat(lo, hi);
}

// ---------------------------------------------------------------------------
// Kernel 1: window-partition gather (faithful to reference raw reshape) + f16
// xt[w*49 + n][c'] = x[b][c][wy*7+iy][wx*7+ix],  f = c'*49 + n,
// iy = f/5376, ix = (f/768)%7, c = f%768
// ---------------------------------------------------------------------------
__global__ __launch_bounds__(256) void gather_x_f16(
    const float* __restrict__ x, _Float16* __restrict__ xt) {
  int idx = blockIdx.x * 256 + threadIdx.x;   // element of [MROWS x DIM]
  int cp  = idx % DIM;
  int row = idx / DIM;
  int w = row / NTOK, n = row % NTOK;
  int b = w >> 6, wy = (w >> 3) & 7, wx = w & 7;
  int f = cp * NTOK + n;
  int iy = f / (7 * DIM);
  int ix = (f / DIM) % 7;
  int c  = f % DIM;
  size_t src = (((size_t)b * DIM + c) * 56 + wy * 7 + iy) * 56 + wx * 7 + ix;
  xt[(size_t)row * DIM + cp] = (_Float16)x[src];
}

__global__ __launch_bounds__(256) void convert_f32_to_f16(
    const float* __restrict__ in, _Float16* __restrict__ out) {
  int idx = blockIdx.x * 256 + threadIdx.x;
  out[idx] = (_Float16)in[idx];
}

// ---------------------------------------------------------------------------
// Kernel 2: WMMA GEMM  C[M,N] = A[M,K] @ W[N,K]^T + bias
// Block tile 128x64, BK=32, 256 threads = 8 waves, each wave 32x32.
// ---------------------------------------------------------------------------
#define BM 128
#define BN 64
#define BK 32
#define LDP 40   // padded LDS row stride (halves); multiple of 8 -> 16B aligned

__global__ __launch_bounds__(256) void gemm_f16_wmma(
    const _Float16* __restrict__ A, const _Float16* __restrict__ W,
    const float* __restrict__ bias, _Float16* __restrict__ out16,
    float* __restrict__ outF, int Nout, int K, int write_f32) {
  __shared__ __align__(16) _Float16 As[BM * LDP];
  __shared__ __align__(16) _Float16 Bs[BN * LDP];

  const int tid = threadIdx.x;
  const int lane = tid & 31;
  const int wv = tid >> 5;
  const int wr = wv >> 1;   // wave row 0..3  (32 rows each)
  const int wc = wv & 1;    // wave col 0..1  (32 cols each)
  const size_t mBase = (size_t)blockIdx.x * BM;
  const size_t nBase = (size_t)blockIdx.y * BN;

  v8f acc[2][2];
#pragma unroll
  for (int mi = 0; mi < 2; ++mi)
#pragma unroll
    for (int ni = 0; ni < 2; ++ni) acc[mi][ni] = (v8f){};

  const int arow = tid >> 1, ach = tid & 1;   // A: 128 rows x (2 x 16h chunks)
  const int brow = tid >> 2, bch = tid & 3;   // B: 64 rows x (4 x 8h chunks)

  for (int k0 = 0; k0 < K; k0 += BK) {
    // stage A tile (128x32) and B tile (64x32) through LDS, 16B vectors
    {
      const v8h* g = (const v8h*)(A + (mBase + arow) * K + k0 + ach * 16);
      v8h x0 = g[0], x1 = g[1];
      *(v8h*)(As + arow * LDP + ach * 16)     = x0;
      *(v8h*)(As + arow * LDP + ach * 16 + 8) = x1;
    }
    {
      const v8h* g = (const v8h*)(W + (nBase + brow) * K + k0 + bch * 8);
      *(v8h*)(Bs + brow * LDP + bch * 8) = g[0];
    }
    if (k0 + BK < K) {  // CDNA5 global_prefetch_b8 for next K tile
      __builtin_prefetch(A + (mBase + arow) * K + k0 + BK, 0, 1);
      __builtin_prefetch(W + (nBase + brow) * K + k0 + BK, 0, 1);
    }
    __syncthreads();

    v16h bfrag[2];
#pragma unroll
    for (int ni = 0; ni < 2; ++ni)
      bfrag[ni] = load_fragB(Bs + (wc * 32 + ni * 16) * LDP, LDP, lane);
#pragma unroll
    for (int mi = 0; mi < 2; ++mi) {
      v16h afrag = load_fragA(As + (wr * 32 + mi * 16) * LDP, LDP, lane);
#pragma unroll
      for (int ni = 0; ni < 2; ++ni)
        acc[mi][ni] = wmma16(afrag, bfrag[ni], acc[mi][ni]);
    }
    __syncthreads();
  }

  // epilogue: C layout row = r + 8*(lane>>4), col = lane&15
  const int hl = lane >> 4, nn = lane & 15;
#pragma unroll
  for (int mi = 0; mi < 2; ++mi)
#pragma unroll
    for (int ni = 0; ni < 2; ++ni)
#pragma unroll
      for (int r = 0; r < 8; ++r) {
        size_t row = mBase + wr * 32 + mi * 16 + r + 8 * hl;
        size_t col = nBase + wc * 32 + ni * 16 + nn;
        float v = acc[mi][ni][r] + bias[col];
        if (write_f32) outF[row * Nout + col] = v;
        else           out16[row * Nout + col] = (_Float16)v;
      }
}

// ---------------------------------------------------------------------------
// Kernel 3: window attention, one wave per (window, head).
// S(49x49 padded 64x64) = (q*scale) k^T + bias + mask; softmax; O = P v.
// ---------------------------------------------------------------------------
__global__ __launch_bounds__(32) void attn_wmma(
    const _Float16* __restrict__ qkv,        // [MROWS, 2304]
    const float* __restrict__ bias_table,    // [169, 24]
    const int* __restrict__ rel_index,       // [49*49]
    const float* __restrict__ mask,          // [64, 49, 49]
    _Float16* __restrict__ out) {            // [MROWS, 768]
  __shared__ __align__(16) _Float16 qs[64 * LDP];
  __shared__ __align__(16) _Float16 ks[64 * LDP];
  __shared__ __align__(16) _Float16 vt[32 * 72];  // V^T: [d][t], pad stride 72
  __shared__ __align__(16) _Float16 ps[16 * 72];  // P row-strip

  const int lane = threadIdx.x;
  const int wh = blockIdx.x;
  const int w = wh / NHEADS, h = wh % NHEADS;
  const int widx = w & 63;
  const float scale = 0.17677669529663687f;   // 1/sqrt(32)
  const size_t tokBase = (size_t)w * NTOK;

  // Load q, k (zero-padded to 64 rows) and v transposed into LDS.
#pragma unroll
  for (int rr = 0; rr < 2; ++rr) {
    int t = rr * 32 + lane;
    if (t < NTOK) {
      const _Float16* src = qkv + (tokBase + t) * QKVDIM + h * HD;
#pragma unroll
      for (int c = 0; c < 4; ++c) {
        *(v8h*)(qs + t * LDP + c * 8) = ((const v8h*)src)[c];
        *(v8h*)(ks + t * LDP + c * 8) = ((const v8h*)(src + DIM))[c];
        v8h vv = ((const v8h*)(src + 2 * DIM))[c];
#pragma unroll
        for (int e = 0; e < 8; ++e) vt[(c * 8 + e) * 72 + t] = vv[e];
      }
    } else {
      v8h z = (v8h){};
#pragma unroll
      for (int c = 0; c < 4; ++c) {
        *(v8h*)(qs + t * LDP + c * 8) = z;
        *(v8h*)(ks + t * LDP + c * 8) = z;
      }
#pragma unroll
      for (int d = 0; d < 32; ++d) vt[d * 72 + t] = (_Float16)0.0f;
    }
  }
  __syncthreads();

  const int hl = lane >> 4, nn = lane & 15;

  for (int mt = 0; mt < 4; ++mt) {
    // S row-strip (16x64): 4 WMMAs, K = 32 = hd exactly.
    v16h qa = load_fragA(qs + (mt * 16) * LDP, LDP, lane);
    v8f s[4];
#pragma unroll
    for (int nt = 0; nt < 4; ++nt) {
      v16h kb = load_fragB(ks + (nt * 16) * LDP, LDP, lane);
      s[nt] = wmma16(qa, kb, (v8f){});
    }

    // scale + relative-position bias + mask, row softmax (rows in lane halves)
#pragma unroll
    for (int r = 0; r < 8; ++r) {
      int i = mt * 16 + r + 8 * hl;           // query token
      float vals[4];
      float vmax = -3.0e38f;
#pragma unroll
      for (int nt = 0; nt < 4; ++nt) {
        int j = nt * 16 + nn;                 // key token
        float v;
        if (i < NTOK && j < NTOK) {
          v = s[nt][r] * scale
            + bias_table[rel_index[i * NTOK + j] * NHEADS + h]
            + mask[(size_t)widx * NTOK * NTOK + i * NTOK + j];
        } else {
          v = -3.0e38f;                       // pad -> softmax weight 0
        }
        vals[nt] = v;
        vmax = fmaxf(vmax, v);
      }
#pragma unroll
      for (int off = 1; off < 16; off <<= 1)
        vmax = fmaxf(vmax, __shfl_xor(vmax, off, 32));
      float sum = 0.0f;
#pragma unroll
      for (int nt = 0; nt < 4; ++nt) {
        float e = __expf(vals[nt] - vmax);
        vals[nt] = e;
        sum += e;
      }
#pragma unroll
      for (int off = 1; off < 16; off <<= 1)
        sum += __shfl_xor(sum, off, 32);
      float inv = 1.0f / sum;
#pragma unroll
      for (int nt = 0; nt < 4; ++nt)
        ps[(r + 8 * hl) * 72 + nt * 16 + nn] = (_Float16)(vals[nt] * inv);
    }
    __syncthreads();

    // O strip (16x32) = P(16x64) @ V(64x32): 2 N-tiles x 2 K-steps.
    v8f o[2] = {(v8f){}, (v8f){}};
#pragma unroll
    for (int kk = 0; kk < 2; ++kk) {
      v16h pa = load_fragA(ps + 32 * kk, 72, lane);
#pragma unroll
      for (int nt2 = 0; nt2 < 2; ++nt2) {
        v16h vb = load_fragB(vt + (nt2 * 16) * 72 + 32 * kk, 72, lane);
        o[nt2] = wmma16(pa, vb, o[nt2]);
      }
    }
#pragma unroll
    for (int nt2 = 0; nt2 < 2; ++nt2)
#pragma unroll
      for (int r = 0; r < 8; ++r) {
        int i = mt * 16 + r + 8 * hl;
        if (i < NTOK)
          out[(tokBase + i) * DIM + h * HD + nt2 * 16 + nn] = (_Float16)o[nt2][r];
      }
    __syncthreads();   // protect ps before next strip
  }
}

// ---------------------------------------------------------------------------
// Launch: gather/convert -> QKV GEMM -> attention -> proj GEMM (f32 -> d_out).
// d_out flat == (B_, N, C) flat per the reference's raw reshape.
// ---------------------------------------------------------------------------
extern "C" void kernel_launch(void* const* d_in, const int* in_sizes, int n_in,
                              void* d_out, int out_size, void* d_ws, size_t ws_size,
                              hipStream_t stream) {
  const float* x          = (const float*)d_in[0];
  const float* mask       = (const float*)d_in[1];
  const float* qkv_w      = (const float*)d_in[2];
  const float* qkv_b      = (const float*)d_in[3];
  const float* proj_w     = (const float*)d_in[4];
  const float* proj_b     = (const float*)d_in[5];
  const float* bias_table = (const float*)d_in[6];
  const int*   rel_index  = (const int*)d_in[7];
  float* out = (float*)d_out;

  char* ws = (char*)d_ws;
  size_t off = 0;
  auto take = [&](size_t bytes) -> char* {
    char* p = ws + off;
    off = (off + bytes + 255) & ~(size_t)255;
    return p;
  };
  _Float16* xt   = (_Float16*)take((size_t)MROWS * DIM * 2);
  _Float16* wq16 = (_Float16*)take((size_t)QKVDIM * DIM * 2);
  _Float16* wp16 = (_Float16*)take((size_t)DIM * DIM * 2);
  _Float16* qkvb = (_Float16*)take((size_t)MROWS * QKVDIM * 2);
  _Float16* ao   = (_Float16*)take((size_t)MROWS * DIM * 2);

  // 1) gather + precision conversion
  gather_x_f16<<<(MROWS * DIM) / 256, 256, 0, stream>>>(x, xt);
  convert_f32_to_f16<<<(QKVDIM * DIM) / 256, 256, 0, stream>>>(qkv_w, wq16);
  convert_f32_to_f16<<<(DIM * DIM) / 256, 256, 0, stream>>>(proj_w, wp16);

  // 2) QKV projection: [50176,768] x [2304,768]^T -> f16 [50176,2304]
  gemm_f16_wmma<<<dim3(MROWS / BM, QKVDIM / BN), 256, 0, stream>>>(
      xt, wq16, qkv_b, qkvb, nullptr, QKVDIM, DIM, 0);

  // 3) windowed multi-head attention: one wave per (window, head)
  attn_wmma<<<NWIN * NHEADS, 32, 0, stream>>>(qkvb, bias_table, rel_index, mask, ao);

  // 4) output projection straight into d_out (flat == NCHW via raw reshape)
  gemm_f16_wmma<<<dim3(MROWS / BM, DIM / BN), 256, 0, stream>>>(
      ao, wp16, proj_b, nullptr, out, DIM, DIM, 1);
}
